// SSMAttentionBlock_37769942401652
// MI455X (gfx1250) — compile-verified, matrix operand order checked by reading, not run on hardware
//
#include <hip/hip_runtime.h>
#include <math.h>

// ---- problem constants (distinct names to avoid macro collisions) ----
static constexpr int kB = 2, kL = 2048, kD = 1024, kH = 16, kHD = 64;
static constexpr int kCAP = 1024, kSINKS = 4;
static constexpr float kEPS = 1e-5f;
static constexpr float kSCALE = 0.125f; // 1/sqrt(64)

typedef __attribute__((ext_vector_type(16))) _Float16 v16h;
typedef __attribute__((ext_vector_type(8)))  _Float16 v8h;
typedef __attribute__((ext_vector_type(4)))  _Float16 v4h;
typedef __attribute__((ext_vector_type(8)))  float    v8f;

// ---------------------------------------------------------------------
// Generic row LayerNorm: rows of `width` contiguous elements.
// ---------------------------------------------------------------------
__global__ void ln_kernel(const float* __restrict__ in, const float* __restrict__ g,
                          const float* __restrict__ bt, float* __restrict__ out, int width)
{
    const int row = blockIdx.x;
    const int tid = threadIdx.x, bd = blockDim.x;
    const float* x = in + (size_t)row * width;
    float* o = out + (size_t)row * width;
    __shared__ float red[256];

    float s = 0.f;
    for (int i = tid; i < width; i += bd) s += x[i];
    red[tid] = s; __syncthreads();
    for (int st = bd >> 1; st > 0; st >>= 1) { if (tid < st) red[tid] += red[tid + st]; __syncthreads(); }
    const float mu = red[0] / (float)width;
    __syncthreads();

    float v = 0.f;
    for (int i = tid; i < width; i += bd) { float d0 = x[i] - mu; v += d0 * d0; }
    red[tid] = v; __syncthreads();
    for (int st = bd >> 1; st > 0; st >>= 1) { if (tid < st) red[tid] += red[tid + st]; __syncthreads(); }
    const float inv = rsqrtf(red[0] / (float)width + kEPS);
    __syncthreads();

    for (int i = tid; i < width; i += bd) o[i] = (x[i] - mu) * inv * g[i] + bt[i];
}

// ---------------------------------------------------------------------
// SSM: FFT conv with h[t]=exp(-alpha t) == first-order recurrence.
// ---------------------------------------------------------------------
__global__ void ssm_kernel(const float* __restrict__ xn, const float* __restrict__ x,
                           const float* __restrict__ log_alpha,
                           float* __restrict__ x1, float* __restrict__ carry)
{
    const int idx = blockIdx.x * blockDim.x + threadIdx.x;
    if (idx >= kB * kD) return;
    const int b = idx / kD, d = idx % kD;

    const float la    = log_alpha[d];
    const float alpha = log1pf(expf(la));        // softplus
    const float r     = expf(-alpha);
    const float r2    = r * r;
    const float geo   = (1.0f - expf(-2.0f * alpha * (float)kL)) / (1.0f - r2);
    const float inv   = 1.0f / fmaxf(sqrtf(geo), 1e-6f);

    const float* xnb = xn + (size_t)b * kL * kD + d;
    const float* xb  = x  + (size_t)b * kL * kD + d;
    float*       ob  = x1 + (size_t)b * kL * kD + d;

    float y = 0.f, yn = 0.f;
    for (int t = 0; t < kL; ++t) {
        y  = xnb[(size_t)t * kD] + r * y;
        yn = y * inv;
        ob[(size_t)t * kD] = xb[(size_t)t * kD] + yn;
    }
    carry[idx] = yn;
}

// ---------------------------------------------------------------------
// Tiled WMMA GEMM:  C[M,N] = act( scale * A[M,K] * B' + bias ) + resid
//   BTRANS==0 : B' = W[N,K]^T    BTRANS==1 : B' = W[K,N]
// Block = 256 threads (8 waves) -> 32(M) x 64(N) tile, K slab of 64.
// A,B staged in LDS as f16 (B stored N-major either way), fragments read
// as two 16B ds_load_b128 per operand, 2 x v_wmma_f32_16x16x32_f16 per slab.
// Requires: M%32==0, N%64==0, K%64==0 (true for every GEMM here).
// ---------------------------------------------------------------------
template<int BTRANS, int ACT>
__global__ __launch_bounds__(256)
void wmma_gemm_tiled(const float* __restrict__ A, const float* __restrict__ W,
                     const float* __restrict__ bias, const float* __restrict__ resid,
                     float* __restrict__ C, int M, int N, int K,
                     int lda, int ldb, int ldc, float scale)
{
    constexpr int BM = 32, BN = 64, BK = 64;
    constexpr int LDT = 72;              // halves; 144B row stride: 16B-aligned, conflict-free
    __shared__ _Float16 As[BM * LDT];    // 32 x 64 (k-major rows)
    __shared__ _Float16 Bs[BN * LDT];    // 64 x 64 (N rows, k-major) -- transposed on stage if BTRANS

    const int tid   = threadIdx.x;
    const int tileM = blockIdx.y * BM;
    const int tileN = blockIdx.x * BN;

    const int wave  = tid >> 5;          // 0..7
    const int wm    = wave >> 2;         // 0..1 -> M sub-tile
    const int wn    = wave & 3;          // 0..3 -> N sub-tile
    const int lane  = tid & 31;
    const int halfq = lane >> 4;         // K sub-block selector
    const int r     = lane & 15;         // A row / B column within 16x16 tile

    v8f acc0 = {};                       // two accumulators? no: one 16x16 tile per wave
    for (int k0 = 0; k0 < K; k0 += BK) {
        __syncthreads();                 // protect LDS from previous slab's readers

        // ---- stage A: 32 rows x 64 k  (512 float4, 2 per thread) ----
        #pragma unroll
        for (int it = 0; it < 2; ++it) {
            const int e   = tid + it * 256;
            const int row = e >> 4;          // 0..31
            const int seg = e & 15;          // float4 index along K
            const float4 f = *reinterpret_cast<const float4*>(
                &A[(size_t)(tileM + row) * lda + k0 + seg * 4]);
            v4h h; h[0] = (_Float16)f.x; h[1] = (_Float16)f.y;
                   h[2] = (_Float16)f.z; h[3] = (_Float16)f.w;
            *reinterpret_cast<v4h*>(&As[row * LDT + seg * 4]) = h;
        }
        // ---- stage B: 64 x 64 (1024 float4, 4 per thread) ----
        #pragma unroll
        for (int it = 0; it < 4; ++it) {
            const int e = tid + it * 256;
            if (BTRANS == 0) {
                const int row = e >> 4;      // N row 0..63
                const int seg = e & 15;      // along K
                const float4 f = *reinterpret_cast<const float4*>(
                    &W[(size_t)(tileN + row) * ldb + k0 + seg * 4]);
                v4h h; h[0] = (_Float16)f.x; h[1] = (_Float16)f.y;
                       h[2] = (_Float16)f.z; h[3] = (_Float16)f.w;
                *reinterpret_cast<v4h*>(&Bs[row * LDT + seg * 4]) = h;
            } else {
                const int krow = e >> 4;     // K row 0..63
                const int seg  = e & 15;     // along N
                const float4 f = *reinterpret_cast<const float4*>(
                    &W[(size_t)(k0 + krow) * ldb + tileN + seg * 4]);
                Bs[(seg * 4 + 0) * LDT + krow] = (_Float16)f.x;
                Bs[(seg * 4 + 1) * LDT + krow] = (_Float16)f.y;
                Bs[(seg * 4 + 2) * LDT + krow] = (_Float16)f.z;
                Bs[(seg * 4 + 3) * LDT + krow] = (_Float16)f.w;
            }
        }
        // prefetch next K slab of A into cache (global_prefetch_b8)
        if (k0 + BK < K)
            __builtin_prefetch(&A[(size_t)(tileM + (tid >> 3)) * lda + k0 + BK], 0, 1);

        __syncthreads();

        // ---- compute: 2 WMMAs over the 64-wide slab ----
        const int ar = (wm * 16 + r) * LDT;
        const int br = (wn * 16 + r) * LDT;
        #pragma unroll
        for (int sub = 0; sub < 2; ++sub) {
            const int ko = sub * 32 + halfq * 8;
            const v8h alo = *reinterpret_cast<const v8h*>(&As[ar + ko]);
            const v8h ahi = *reinterpret_cast<const v8h*>(&As[ar + ko + 16]);
            const v8h blo = *reinterpret_cast<const v8h*>(&Bs[br + ko]);
            const v8h bhi = *reinterpret_cast<const v8h*>(&Bs[br + ko + 16]);
            const v16h af = __builtin_shufflevector(alo, ahi, 0,1,2,3,4,5,6,7,8,9,10,11,12,13,14,15);
            const v16h bf = __builtin_shufflevector(blo, bhi, 0,1,2,3,4,5,6,7,8,9,10,11,12,13,14,15);
            acc0 = __builtin_amdgcn_wmma_f32_16x16x32_f16(
                       false, af, false, bf, (short)0, acc0, false, false);
        }
    }

    // ---- epilogue ----
    #pragma unroll
    for (int i = 0; i < 8; ++i) {
        const int row = tileM + wm * 16 + i + halfq * 8;
        const int col = tileN + wn * 16 + r;
        float v = acc0[i] * scale;
        if (bias) v += bias[col];
        if (ACT)  v = 0.5f * v * (1.0f + erff(v * 0.70710678118654752f)); // exact gelu
        if (resid) v += resid[(size_t)row * ldc + col];
        C[(size_t)row * ldc + col] = v;
    }
}

// ---------------------------------------------------------------------
// RoPE in-place on (B,L,H*HD) layout; one thread per rotation pair.
// ---------------------------------------------------------------------
__global__ void rope_kernel(float* __restrict__ x, const float* __restrict__ freqs)
{
    const int total = kB * kL * kH * (kHD / 2);
    int i = blockIdx.x * blockDim.x + threadIdx.x;
    if (i >= total) return;
    const int d = i % (kHD / 2); i /= (kHD / 2);
    const int h = i % kH;        i /= kH;
    const int l = i % kL;        const int b = i / kL;

    const size_t base = ((size_t)b * kL + l) * kD + h * kHD + 2 * d;
    const float a  = x[base], bb = x[base + 1];
    const float c  = freqs[((size_t)l * (kHD / 2) + d) * 2 + 0];
    const float s  = freqs[((size_t)l * (kHD / 2) + d) * 2 + 1];
    x[base]     = a * c - bb * s;
    x[base + 1] = a * s + bb * c;
}

// ---------------------------------------------------------------------
// Saliency pass 1: per-row (b,l) softmax stats of head-0 causal scores.
// ---------------------------------------------------------------------
__global__ void sal_rowstats_kernel(const float* __restrict__ q, const float* __restrict__ k,
                                    float* __restrict__ rowmax, float* __restrict__ rowsum)
{
    const int l = blockIdx.x, b = blockIdx.y, tid = threadIdx.x;
    const float* qr = q + ((size_t)b * kL + l) * kD;   // head 0
    const float* kb = k + (size_t)b * kL * kD;
    __shared__ float red[128];

    float m = -1e30f;
    for (int s = tid; s <= l; s += 128) {
        const float* kr = kb + (size_t)s * kD;
        float dot = 0.f;
        for (int d = 0; d < kHD; ++d) dot += qr[d] * kr[d];
        m = fmaxf(m, dot * kSCALE);
    }
    red[tid] = m; __syncthreads();
    for (int st = 64; st > 0; st >>= 1) { if (tid < st) red[tid] = fmaxf(red[tid], red[tid + st]); __syncthreads(); }
    const float rm = red[0]; __syncthreads();

    float s0 = 0.f;
    for (int s = tid; s <= l; s += 128) {
        const float* kr = kb + (size_t)s * kD;
        float dot = 0.f;
        for (int d = 0; d < kHD; ++d) dot += qr[d] * kr[d];
        s0 += expf(dot * kSCALE - rm);
    }
    red[tid] = s0; __syncthreads();
    for (int st = 64; st > 0; st >>= 1) { if (tid < st) red[tid] += red[tid + st]; __syncthreads(); }
    if (tid == 0) { rowmax[b * kL + l] = rm; rowsum[b * kL + l] = red[0]; }
}

// Saliency pass 2: sal[b,s] = sum_{l>=s} P[l,s]  (deterministic, no atomics)
__global__ void sal_col_kernel(const float* __restrict__ q, const float* __restrict__ k,
                               const float* __restrict__ rowmax, const float* __restrict__ rowsum,
                               float* __restrict__ sal)
{
    const int s = blockIdx.x, b = blockIdx.y, tid = threadIdx.x;
    const float* ks = k + ((size_t)b * kL + s) * kD;   // head 0
    const float* qb = q + (size_t)b * kL * kD;
    __shared__ float red[128];

    float acc = 0.f;
    for (int l = s + tid; l < kL; l += 128) {
        const float* qr = qb + (size_t)l * kD;
        float dot = 0.f;
        for (int d = 0; d < kHD; ++d) dot += qr[d] * ks[d];
        acc += expf(dot * kSCALE - rowmax[b * kL + l]) / rowsum[b * kL + l];
    }
    red[tid] = acc; __syncthreads();
    for (int st = 64; st > 0; st >>= 1) { if (tid < st) red[tid] += red[tid + st]; __syncthreads(); }
    if (tid == 0) sal[b * kL + s] = red[0];
}

// ---------------------------------------------------------------------
// Top-k by rank counting (matches jax top_k tie-break: lower index wins).
// ---------------------------------------------------------------------
__global__ void topk_kernel(const float* __restrict__ sal, int* __restrict__ keep)
{
    const int b = blockIdx.x, tid = threadIdx.x, bd = blockDim.x;
    const float* s_ = sal + (size_t)b * kL;
    __shared__ int flags[kL];
    __shared__ int pos[kL];

    for (int i = tid; i < kL; i += bd) {
        int f = 0;
        if (i < kSINKS) f = 1;
        else {
            const float vi = s_[i];
            int rank = 0;
            for (int j = kSINKS; j < kL; ++j) {
                if (j == i) continue;
                const float vj = s_[j];
                if (vj > vi || (vj == vi && j < i)) ++rank;
            }
            f = (rank < (kCAP - kSINKS)) ? 1 : 0;
        }
        flags[i] = f;
    }
    __syncthreads();
    if (tid == 0) { int c = 0; for (int i = 0; i < kL; ++i) { pos[i] = c; c += flags[i]; } }
    __syncthreads();
    for (int i = tid; i < kL; i += bd)
        if (flags[i]) keep[b * kCAP + pos[i]] = i;
}

// Gather pruned KV (+ se) into the output layout (B,H,CAP,HD).
__global__ void gather_kernel(const float* __restrict__ k, const float* __restrict__ v,
                              const float* __restrict__ sal, const int* __restrict__ keep,
                              float* __restrict__ ke, float* __restrict__ ve, float* __restrict__ se)
{
    const int total = kB * kH * kCAP * kHD;
    int t = blockIdx.x * blockDim.x + threadIdx.x;
    if (t >= total) return;
    const int d = t % kHD; int r = t / kHD;
    const int s = r % kCAP; r /= kCAP;
    const int h = r % kH;   const int b = r / kH;

    const int src = keep[b * kCAP + s];
    const size_t so = ((size_t)b * kL + src) * kD + h * kHD + d;
    ke[t] = k[so];
    ve[t] = v[so];
    if (h == 0 && d == 0) se[b * kCAP + s] = sal[(size_t)b * kL + src];
}

// Causal(ish) softmax over P rows; positions s >= min(l+1,CAP) zeroed.
__global__ void attn_softmax_kernel(float* __restrict__ P)
{
    const int l = blockIdx.x, tid = threadIdx.x;
    float* row = P + (size_t)l * kCAP;
    const int valid = (l + 1 < kCAP) ? (l + 1) : kCAP;
    __shared__ float red[256];

    float m = -1e30f;
    for (int i = tid; i < valid; i += 256) m = fmaxf(m, row[i]);
    red[tid] = m; __syncthreads();
    for (int st = 128; st > 0; st >>= 1) { if (tid < st) red[tid] = fmaxf(red[tid], red[tid + st]); __syncthreads(); }
    const float rm = red[0]; __syncthreads();

    float s = 0.f;
    for (int i = tid; i < valid; i += 256) s += expf(row[i] - rm);
    red[tid] = s; __syncthreads();
    for (int st = 128; st > 0; st >>= 1) { if (tid < st) red[tid] += red[tid + st]; __syncthreads(); }
    const float inv = 1.0f / red[0]; __syncthreads();

    for (int i = tid; i < kCAP; i += 256)
        row[i] = (i < valid) ? expf(row[i] - rm) * inv : 0.0f;
}

// ---------------------------------------------------------------------
extern "C" void kernel_launch(void* const* d_in, const int* in_sizes, int n_in,
                              void* d_out, int out_size, void* d_ws, size_t ws_size,
                              hipStream_t stream)
{
    (void)in_sizes; (void)n_in; (void)out_size; (void)ws_size;

    const float* x         = (const float*)d_in[0];
    const float* freqs     = (const float*)d_in[1];
    const float* log_alpha = (const float*)d_in[2];
    const float* g_ssm     = (const float*)d_in[3];
    const float* b_ssm     = (const float*)d_in[4];
    const float* g_attn    = (const float*)d_in[5];
    const float* b_attn    = (const float*)d_in[6];
    const float* wq        = (const float*)d_in[7];
    const float* wk        = (const float*)d_in[8];
    const float* wv        = (const float*)d_in[9];
    const float* wo        = (const float*)d_in[10];
    const float* gq        = (const float*)d_in[11];
    const float* bq        = (const float*)d_in[12];
    const float* gk        = (const float*)d_in[13];
    const float* bk        = (const float*)d_in[14];
    const float* g_mlp     = (const float*)d_in[15];
    const float* b_mlp     = (const float*)d_in[16];
    const float* w1        = (const float*)d_in[17];
    const float* b1        = (const float*)d_in[18];
    const float* w2        = (const float*)d_in[19];
    const float* b2        = (const float*)d_in[20];

    const size_t X = (size_t)kB * kL * kD;           // 4,194,304
    float* out      = (float*)d_out;
    float* out_x    = out;                            // (B,L,D)
    float* out_cy   = out + X;                        // (B,D)
    float* out_ke   = out_cy + (size_t)kB * kD;       // (B,H,CAP,HD)
    float* out_ve   = out_ke + (size_t)kB * kH * kCAP * kHD;
    float* out_se   = out_ve + (size_t)kB * kH * kCAP * kHD;   // (B,CAP)

    float* ws = (float*)d_ws;
    float* xn = ws;  ws += X;     // LN(ssm) output, reused later as MLP-LN
    float* x1 = ws;  ws += X;     // x after SSM residual
    float* hh = ws;  ws += X;     // LN(attn) output
    float* q  = ws;  ws += X;
    float* k  = ws;  ws += X;
    float* v  = ws;  ws += X;
    float* ao = ws;  ws += X;     // attention output (B,L,D)
    float* x2 = ws;  ws += X;     // x after attention residual
    float* h1 = ws;  ws += 4 * X; // MLP hidden (B,L,4D)
    float* P  = ws;  ws += (size_t)kL * kCAP;         // per-(b,h) score/prob buffer
    float* rowmax = ws; ws += (size_t)kB * kL;
    float* rowsum = ws; ws += (size_t)kB * kL;
    float* sal    = ws; ws += (size_t)kB * kL;
    int*   keep   = (int*)ws;

    // 1) SSM branch
    ln_kernel<<<kB * kL, 256, 0, stream>>>(x, g_ssm, b_ssm, xn, kD);
    ssm_kernel<<<(kB * kD + 255) / 256, 256, 0, stream>>>(xn, x, log_alpha, x1, out_cy);

    // 2) Attention branch: LN + QKV projections (WMMA)
    ln_kernel<<<kB * kL, 256, 0, stream>>>(x1, g_attn, b_attn, hh, kD);
    {
        dim3 g(kD / 64, (kB * kL) / 32);
        wmma_gemm_tiled<0,0><<<g, 256, 0, stream>>>(hh, wq, nullptr, nullptr, q,
                                                    kB * kL, kD, kD, kD, kD, kD, 1.0f);
        wmma_gemm_tiled<0,0><<<g, 256, 0, stream>>>(hh, wk, nullptr, nullptr, k,
                                                    kB * kL, kD, kD, kD, kD, kD, 1.0f);
        wmma_gemm_tiled<0,0><<<g, 256, 0, stream>>>(hh, wv, nullptr, nullptr, v,
                                                    kB * kL, kD, kD, kD, kD, kD, 1.0f);
    }

    // per-head q/k LayerNorm (rows of 64 contiguous floats), then RoPE
    ln_kernel<<<kB * kL * kH, 64, 0, stream>>>(q, gq, bq, q, kHD);
    ln_kernel<<<kB * kL * kH, 64, 0, stream>>>(k, gk, bk, k, kHD);
    {
        const int total = kB * kL * kH * (kHD / 2);
        rope_kernel<<<(total + 255) / 256, 256, 0, stream>>>(q, freqs);
        rope_kernel<<<(total + 255) / 256, 256, 0, stream>>>(k, freqs);
    }

    // 3) Head-0 saliency + top-k KV pruning + gather (ke/ve/se are outputs)
    sal_rowstats_kernel<<<dim3(kL, kB), 128, 0, stream>>>(q, k, rowmax, rowsum);
    sal_col_kernel<<<dim3(kL, kB), 128, 0, stream>>>(q, k, rowmax, rowsum, sal);
    topk_kernel<<<kB, 256, 0, stream>>>(sal, keep);
    {
        const int total = kB * kH * kCAP * kHD;
        gather_kernel<<<(total + 255) / 256, 256, 0, stream>>>(k, v, sal, keep,
                                                               out_ke, out_ve, out_se);
    }

    // 4) Pruned attention, serialized per (b,h): scores(WMMA) -> softmax -> PV(WMMA)
    for (int b = 0; b < kB; ++b) {
        for (int h = 0; h < kH; ++h) {
            const float* qbh = q + (size_t)b * kL * kD + h * kHD;                 // (L,HD) ld=D
            const float* kebh = out_ke + ((size_t)b * kH + h) * kCAP * kHD;       // (CAP,HD)
            const float* vebh = out_ve + ((size_t)b * kH + h) * kCAP * kHD;       // (CAP,HD)
            float* aobh = ao + (size_t)b * kL * kD + h * kHD;                     // (L,HD) ld=D

            dim3 gs(kCAP / 64, kL / 32);
            wmma_gemm_tiled<0,0><<<gs, 256, 0, stream>>>(qbh, kebh, nullptr, nullptr, P,
                                                         kL, kCAP, kHD, kD, kHD, kCAP, kSCALE);
            attn_softmax_kernel<<<kL, 256, 0, stream>>>(P);
            dim3 gp(kHD / 64, kL / 32);
            wmma_gemm_tiled<1,0><<<gp, 256, 0, stream>>>(P, vebh, nullptr, nullptr, aobh,
                                                         kL, kHD, kCAP, kCAP, kHD, kD, 1.0f);
        }
    }

    // 5) Output projection with residual: x2 = x1 + ao @ wo^T
    {
        dim3 g(kD / 64, (kB * kL) / 32);
        wmma_gemm_tiled<0,0><<<g, 256, 0, stream>>>(ao, wo, nullptr, x1, x2,
                                                    kB * kL, kD, kD, kD, kD, kD, 1.0f);
    }

    // 6) MLP: out_x = x2 + gelu(LN(x2) @ w1 + b1) @ w2 + b2   (w1:(D,4D), w2:(4D,D))
    ln_kernel<<<kB * kL, 256, 0, stream>>>(x2, g_mlp, b_mlp, xn, kD);
    {
        dim3 g1(4 * kD / 64, (kB * kL) / 32);
        wmma_gemm_tiled<1,1><<<g1, 256, 0, stream>>>(xn, w1, b1, nullptr, h1,
                                                     kB * kL, 4 * kD, kD, kD, 4 * kD, 4 * kD, 1.0f);
        dim3 g2(kD / 64, (kB * kL) / 32);
        wmma_gemm_tiled<1,0><<<g2, 256, 0, stream>>>(h1, w2, b2, x2, out_x,
                                                     kB * kL, kD, 4 * kD, 4 * kD, kD, kD, 1.0f);
    }
}
